// MultiHeadAttention_53343493816663
// MI455X (gfx1250) — matrix operand order repaired and checked by reading, not compiled, not run on hardware
//
#include <hip/hip_runtime.h>
#include <math.h>

// ---- problem constants (from reference) ----
#define E_DIM   1024
#define HEADS   16
#define HD      64
#define BATCH   4
#define SEQ     2048
#define M_TOTAL (BATCH * SEQ)   // 8192 rows

typedef _Float16 v16h __attribute__((ext_vector_type(16)));
typedef _Float16 v8h  __attribute__((ext_vector_type(8)));
typedef float    v8f  __attribute__((ext_vector_type(8)));

union H16 { v16h v; v8h h[2]; };

// D = A(16x32 f16) * B(32x16 f16) + C(16x16 f32)
__device__ __forceinline__ v8f wmma_f16(v16h a, v16h b, v8f c) {
  return __builtin_amdgcn_wmma_f32_16x16x32_f16(
      /*neg_a=*/false, a, /*neg_b=*/false, b,
      /*c_mod=*/(short)0, c, /*reuse_a=*/false, /*reuse_b=*/false);
}

// B-frag: lane holds column n = lane&15; halfs 0-7 = K kb..kb+7, halfs 8-15 = K kb+16..kb+23,
// kb = (lane>>4)*8 (caller adds it); memory contiguous in K for that column.
__device__ __forceinline__ v16h load_b_frag(const _Float16* p) {
  H16 u;
  u.h[0] = *(const v8h*)(p);
  u.h[1] = *(const v8h*)(p + 16);
  return u.v;
}

// A-frag: lane holds row m = lane&15; halfs 0-7 = K kb..kb+7, halfs 8-15 = K kb+16..kb+23,
// kb = (lane>>4)*8 (caller adds it).
__device__ __forceinline__ v16h load_a_frag_f16(const _Float16* row, int kb) {
  H16 u;
  u.h[0] = *(const v8h*)(row + kb);
  u.h[1] = *(const v8h*)(row + kb + 16);
  return u.v;
}

__device__ __forceinline__ v16h load_a_frag_f32(const float* row, int kb) {
  H16 u;
#pragma unroll
  for (int i = 0; i < 8; ++i) u.v[i] = (_Float16)row[kb + i];
#pragma unroll
  for (int i = 0; i < 8; ++i) u.v[8 + i] = (_Float16)row[kb + 16 + i];
  return u.v;
}

// ---------------------------------------------------------------------------
// Kernel 1: W[in][out] f32  ->  Wt[out][in] f16  (so WMMA-B loads are contiguous)
// ---------------------------------------------------------------------------
__global__ void convert_w_kernel(const float* __restrict__ W, _Float16* __restrict__ Wt) {
  int gid = blockIdx.x * blockDim.x + threadIdx.x;   // E*E threads
  int o = gid >> 10;          // / E_DIM
  int i = gid & (E_DIM - 1);
  Wt[(size_t)o * E_DIM + i] = (_Float16)W[(size_t)i * E_DIM + o];
}

// ---------------------------------------------------------------------------
// Kernel 2: projection GEMM  out = X(f32) @ W + b  -> f16, head-major layouts.
// block = 128 threads (4 waves); each wave computes a 16x128 tile (8 WMMA accs),
// all 4 waves share the same n-range so B-frag loads hit the WGP cache.
// out_mode 0: [B,H,S,D] (Q, K)     out_mode 1: [B,H,D,S] (V transposed)
// ---------------------------------------------------------------------------
#define PROJ_NT 8   // 8 x 16 = 128 output columns per wave

__global__ void proj_kernel(const float* __restrict__ X, const _Float16* __restrict__ Wt,
                            const float* __restrict__ bias, _Float16* __restrict__ out,
                            int out_mode) {
  const int lane = threadIdx.x & 31;
  const int wave = threadIdx.x >> 5;
  const int m0 = blockIdx.x * 64 + wave * 16;
  const int n0 = blockIdx.y * (16 * PROJ_NT);
  const int hi  = lane >> 4;     // which 16-lane half
  const int l16 = lane & 15;

  v8f acc[PROJ_NT] = {};
  const float* arow = X + (size_t)(m0 + l16) * E_DIM;
  const int ak = hi * 8;

  for (int k = 0; k < E_DIM; k += 32) {
    v16h a = load_a_frag_f32(arow, k + ak);
#pragma unroll
    for (int nt = 0; nt < PROJ_NT; ++nt) {
      const int n = n0 + nt * 16 + l16;
      v16h b = load_b_frag(Wt + (size_t)n * E_DIM + k + hi * 8);
      acc[nt] = wmma_f16(a, b, acc[nt]);
    }
  }

#pragma unroll
  for (int nt = 0; nt < PROJ_NT; ++nt) {
    const int n = n0 + nt * 16 + l16;
    const float bv = bias[n];
    const int h = n >> 6;        // n / HD
    const int d = n & (HD - 1);
#pragma unroll
    for (int r = 0; r < 8; ++r) {
      const int m  = m0 + r + hi * 8;
      const int b_ = m >> 11;            // m / SEQ
      const int s  = m & (SEQ - 1);
      float val = acc[nt][r] + bv;
      size_t idx;
      if (out_mode == 0) idx = (((size_t)(b_ * HEADS + h)) * SEQ + s) * HD + d;
      else               idx = (((size_t)(b_ * HEADS + h)) * HD + d) * SEQ + s;
      out[idx] = (_Float16)val;
    }
  }
}

// ---------------------------------------------------------------------------
// Kernel 3: flash attention. One wave = one (b,h) and 16 query rows.
// Q,K: [B,H,S,D] f16; Vt: [B,H,D,S] f16; O: [B,S,E] f16.
// ---------------------------------------------------------------------------
__global__ void attn_kernel(const _Float16* __restrict__ Q, const _Float16* __restrict__ K,
                            const _Float16* __restrict__ Vt, _Float16* __restrict__ O) {
  __shared__ __align__(32) _Float16 Pbuf[4][16][32];   // per-wave P tile (16 q x 32 k)

  const int lane = threadIdx.x & 31;
  const int wave = threadIdx.x >> 5;
  const int bh = blockIdx.x;                       // b*HEADS + h
  const int q0 = blockIdx.y * 64 + wave * 16;
  const int hi  = lane >> 4;
  const int l16 = lane & 15;

  const size_t base = (size_t)bh * SEQ * HD;       // also = bh * HD * SEQ for Vt
  const _Float16* qrow = Q + base + (size_t)(q0 + l16) * HD;
  const v16h aq0 = load_a_frag_f16(qrow, hi * 8);        // d 0..31
  const v16h aq1 = load_a_frag_f16(qrow, 32 + hi * 8);   // d 32..63

  float mrow[8], lrow[8];
#pragma unroll
  for (int r = 0; r < 8; ++r) { mrow[r] = -__builtin_inff(); lrow[r] = 0.0f; }
  v8f o[4] = {};
  const float scale = 0.125f;                      // 1/sqrt(64)

  for (int j = 0; j < SEQ; j += 32) {
    // two 16x16 score tiles for keys j..j+31
    v8f s[2];
#pragma unroll
    for (int t = 0; t < 2; ++t) {
      const _Float16* krow = K + base + (size_t)(j + t * 16 + l16) * HD;
      v16h bk0 = load_b_frag(krow + hi * 8);        // d 0..31
      v16h bk1 = load_b_frag(krow + 32 + hi * 8);   // d 32..63
      v8f z = {};
      z = wmma_f16(aq0, bk0, z);
      s[t] = wmma_f16(aq1, bk1, z);
    }

    // online softmax update; P tile to LDS in [row][key] order
#pragma unroll
    for (int r = 0; r < 8; ++r) {
      float a0 = s[0][r] * scale;
      float a1 = s[1][r] * scale;
      float t = fmaxf(a0, a1);
#pragma unroll
      for (int off = 1; off < 16; off <<= 1) t = fmaxf(t, __shfl_xor(t, off, 32));
      const float mnew = fmaxf(mrow[r], t);
      const float p0 = __expf(a0 - mnew);
      const float p1 = __expf(a1 - mnew);
      float psum = p0 + p1;
#pragma unroll
      for (int off = 1; off < 16; off <<= 1) psum += __shfl_xor(psum, off, 32);
      const float corr = __expf(mrow[r] - mnew);
      lrow[r] = lrow[r] * corr + psum;
      mrow[r] = mnew;
#pragma unroll
      for (int nt = 0; nt < 4; ++nt) o[nt][r] *= corr;
      const int row = r + hi * 8;
      Pbuf[wave][row][l16]      = (_Float16)p0;
      Pbuf[wave][row][l16 + 16] = (_Float16)p1;
    }
    __syncthreads();   // uniform trip count across all 4 waves

    // re-load P in A-frag layout
    H16 up;
    up.h[0] = *(const v8h*)&Pbuf[wave][l16][hi * 8];
    up.h[1] = *(const v8h*)&Pbuf[wave][l16][hi * 8 + 16];
    const v16h ap = up.v;

#pragma unroll
    for (int nt = 0; nt < 4; ++nt) {
      const int d = nt * 16 + l16;
      v16h bv = load_b_frag(Vt + base + (size_t)d * SEQ + j + hi * 8);
      o[nt] = wmma_f16(ap, bv, o[nt]);
    }
    __syncthreads();
  }

  // normalize and write O[B,S,E] f16
  const int b_ = bh >> 4;       // bh / HEADS
  const int h  = bh & (HEADS - 1);
#pragma unroll
  for (int nt = 0; nt < 4; ++nt) {
#pragma unroll
    for (int r = 0; r < 8; ++r) {
      const int q = q0 + r + hi * 8;
      const int d = nt * 16 + l16;
      float val = o[nt][r] / lrow[r];
      O[((size_t)b_ * SEQ + q) * E_DIM + h * HD + d] = (_Float16)val;
    }
  }
}

// ---------------------------------------------------------------------------
// Kernel 4: final GEMM  out(f32) = O16(f16 [M][E]) @ Wo + bo
// wave tile 16x128, same sharing scheme as proj_kernel.
// ---------------------------------------------------------------------------
__global__ void out_gemm_kernel(const _Float16* __restrict__ A, const _Float16* __restrict__ Wt,
                                const float* __restrict__ bias, float* __restrict__ out) {
  const int lane = threadIdx.x & 31;
  const int wave = threadIdx.x >> 5;
  const int m0 = blockIdx.x * 64 + wave * 16;
  const int n0 = blockIdx.y * (16 * PROJ_NT);
  const int hi  = lane >> 4;
  const int l16 = lane & 15;

  v8f acc[PROJ_NT] = {};
  const _Float16* arow = A + (size_t)(m0 + l16) * E_DIM;
  const int ak = hi * 8;

  for (int k = 0; k < E_DIM; k += 32) {
    v16h a = load_a_frag_f16(arow, k + ak);
#pragma unroll
    for (int nt = 0; nt < PROJ_NT; ++nt) {
      const int n = n0 + nt * 16 + l16;
      v16h b = load_b_frag(Wt + (size_t)n * E_DIM + k + hi * 8);
      acc[nt] = wmma_f16(a, b, acc[nt]);
    }
  }

#pragma unroll
  for (int nt = 0; nt < PROJ_NT; ++nt) {
    const int n = n0 + nt * 16 + l16;
    const float bv = bias[n];
#pragma unroll
    for (int r = 0; r < 8; ++r) {
      const int m = m0 + r + hi * 8;
      out[(size_t)m * E_DIM + n] = acc[nt][r] + bv;
    }
  }
}

// ---------------------------------------------------------------------------
extern "C" void kernel_launch(void* const* d_in, const int* in_sizes, int n_in,
                              void* d_out, int out_size, void* d_ws, size_t ws_size,
                              hipStream_t stream) {
  const float* value = (const float*)d_in[0];
  const float* key   = (const float*)d_in[1];
  const float* query = (const float*)d_in[2];
  const float* Wv = (const float*)d_in[3];
  const float* bv = (const float*)d_in[4];
  const float* Wk = (const float*)d_in[5];
  const float* bk = (const float*)d_in[6];
  const float* Wq = (const float*)d_in[7];
  const float* bq = (const float*)d_in[8];
  const float* Wo = (const float*)d_in[9];
  const float* bo = (const float*)d_in[10];
  float* out = (float*)d_out;

  char* ws = (char*)d_ws;
  const size_t WSZ = (size_t)E_DIM * E_DIM * sizeof(_Float16);   //  2 MB
  const size_t TSZ = (size_t)M_TOTAL * E_DIM * sizeof(_Float16); // 16 MB
  _Float16* WtQ  = (_Float16*)(ws);
  _Float16* WtK  = (_Float16*)(ws + WSZ);
  _Float16* WtV  = (_Float16*)(ws + 2 * WSZ);
  _Float16* WtO  = (_Float16*)(ws + 3 * WSZ);
  _Float16* Q16  = (_Float16*)(ws + 4 * WSZ);
  _Float16* K16  = (_Float16*)(ws + 4 * WSZ + TSZ);
  _Float16* Vt16 = (_Float16*)(ws + 4 * WSZ + 2 * TSZ);
  _Float16* O16  = (_Float16*)(ws + 4 * WSZ + 3 * TSZ);

  // 1) weights -> f16 transposed
  dim3 cwGrid((E_DIM * E_DIM) / 256);
  convert_w_kernel<<<cwGrid, 256, 0, stream>>>(Wq, WtQ);
  convert_w_kernel<<<cwGrid, 256, 0, stream>>>(Wk, WtK);
  convert_w_kernel<<<cwGrid, 256, 0, stream>>>(Wv, WtV);
  convert_w_kernel<<<cwGrid, 256, 0, stream>>>(Wo, WtO);

  // 2) projections: grid covers (64 rows) x (128 cols) per block
  dim3 pGrid(M_TOTAL / 64, E_DIM / (16 * PROJ_NT));
  proj_kernel<<<pGrid, 128, 0, stream>>>(query, WtQ, bq, Q16, 0);
  proj_kernel<<<pGrid, 128, 0, stream>>>(key,   WtK, bk, K16, 0);
  proj_kernel<<<pGrid, 128, 0, stream>>>(value, WtV, bv, Vt16, 1);

  // 3) attention
  dim3 aGrid(BATCH * HEADS, SEQ / 64);
  attn_kernel<<<aGrid, 128, 0, stream>>>(Q16, K16, Vt16, O16);

  // 4) output projection (writes all of d_out)
  out_gemm_kernel<<<pGrid, 128, 0, stream>>>(O16, WtO, bo, out);
}
